// LSTMClassifier_85040352460848
// MI455X (gfx1250) — compile-verified, matrix operand order checked by reading, not run on hardware
//
#include <hip/hip_runtime.h>
#include <stdint.h>

typedef __attribute__((ext_vector_type(16))) _Float16 v16h;
typedef __attribute__((ext_vector_type(8)))  _Float16 v8h;
typedef __attribute__((ext_vector_type(8)))  float    v8f;
typedef __attribute__((ext_vector_type(4)))  float    f32x4;

#define HID   128
#define GATES 512          // 4*H
#define SEQT  512
#define TLOG  9
#define BATCH 128
#define MROWS 65536        // B*T

__device__ __forceinline__ float fast_sigmoid(float x) {
    x = fminf(fmaxf(x, -30.f), 30.f);
    return 1.0f / (1.0f + __expf(-x));
}
__device__ __forceinline__ float fast_tanh(float x) {
    x = fminf(fmaxf(x, -15.f), 15.f);
    float e = __expf(2.0f * x);
    return (e - 1.0f) / (e + 1.0f);
}

// Pack 8 fp32 (two f32x4 regs) into 8 f16 -- placed at ds_store time so the
// s_wait_loadcnt for the raw loads lands *after* the WMMAs.
__device__ __forceinline__ v8h pack8(f32x4 x, f32x4 y) {
    v8h h;
    h[0] = (_Float16)x[0]; h[1] = (_Float16)x[1];
    h[2] = (_Float16)x[2]; h[3] = (_Float16)x[3];
    h[4] = (_Float16)y[0]; h[5] = (_Float16)y[1];
    h[6] = (_Float16)y[2]; h[7] = (_Float16)y[3];
    return h;
}

// 32-bit LDS byte offset for async-to-LDS VDST operand (addrspace(3) cast).
__device__ __forceinline__ uint32_t lds_addr32(const void* p) {
    return (uint32_t)(uintptr_t)(__attribute__((address_space(3))) const void*)p;
}

// ---------------------------------------------------------------------------
// GEMM (fp32 inputs): Out[t,b,512] = A[M,K]*W[512,K]^T + bias0 + bias1.
// A rows are m = b*T + t; output row is t*B + b (recurrence-friendly layout).
// 256 thr / 8 waves, 64Mx64N, K-step 32, LDS double-buffered; raw f32 loads
// stay in regs across the WMMAs and are packed to f16 at store time.
// ---------------------------------------------------------------------------
__global__ __launch_bounds__(256) void wmma_gemm_bias_f32(
    const float* __restrict__ A, const float* __restrict__ W,
    const float* __restrict__ bias0, const float* __restrict__ bias1,
    float* __restrict__ Out, int K)
{
    __shared__ _Float16 sA[2][4 * 512];
    __shared__ _Float16 sB[2][4 * 512];

    const int tid  = threadIdx.x;
    const int wave = tid >> 5;
    const int lane = tid & 31;
    const int bm   = blockIdx.x * 64;
    const int bn   = blockIdx.y * 64;
    const int wm   = wave & 3;
    const int wn   = (wave >> 2) * 2;

    // thread owns panel row r (0..63) and 8-wide contiguous k-run q (0..3)
    const int r    = tid >> 2;
    const int q    = tid & 3;
    const int hi   = (q & 1) ? 16 : 0;
    const int sb   = (q >> 1) * 8;
    const int ldso = (r >> 4) * 512 + ((r & 15) + hi) * 16 + sb;
    const float* Ap = A + (size_t)(bm + r) * K + q * 8;
    const float* Wp = W + (size_t)(bn + r) * K + q * 8;

    v8f acc0 = {}, acc1 = {};
    const int nIter = K >> 5;

    f32x4 a0 = *(const f32x4*)Ap, a1 = *(const f32x4*)(Ap + 4);
    f32x4 w0 = *(const f32x4*)Wp, w1 = *(const f32x4*)(Wp + 4);
    *(v8h*)&sA[0][ldso] = pack8(a0, a1);
    *(v8h*)&sB[0][ldso] = pack8(w0, w1);

    for (int i = 0; i < nIter; ++i) {
        __syncthreads();
        const int cur = i & 1;
        if (i + 1 < nIter) {            // issue next panel's raw loads
            const float* pa = Ap + (size_t)(i + 1) * 32;
            const float* pw = Wp + (size_t)(i + 1) * 32;
            a0 = *(const f32x4*)pa; a1 = *(const f32x4*)(pa + 4);
            w0 = *(const f32x4*)pw; w1 = *(const f32x4*)(pw + 4);
        }
        v16h af  = *(const v16h*)&sA[cur][wm * 512 + lane * 16];
        v16h bf0 = *(const v16h*)&sB[cur][(wn + 0) * 512 + lane * 16];
        v16h bf1 = *(const v16h*)&sB[cur][(wn + 1) * 512 + lane * 16];
        acc0 = __builtin_amdgcn_wmma_f32_16x16x32_f16(false, af, false, bf0,
                                                      (short)0, acc0, false, false);
        acc1 = __builtin_amdgcn_wmma_f32_16x16x32_f16(false, af, false, bf1,
                                                      (short)0, acc1, false, false);
        if (i + 1 < nIter) {            // cvt + fill other buffer (post-WMMA)
            *(v8h*)&sA[cur ^ 1][ldso] = pack8(a0, a1);
            *(v8h*)&sB[cur ^ 1][ldso] = pack8(w0, w1);
        }
    }

    const int mrow = (lane >= 16) ? 8 : 0;
    const int ncol = lane & 15;
    const int n0   = bn + wn * 16 + ncol;
    const int n1   = n0 + 16;
    const float bb0 = bias0[n0] + bias1[n0];
    const float bb1 = bias0[n1] + bias1[n1];
    #pragma unroll
    for (int v = 0; v < 8; ++v) {
        int m = bm + wm * 16 + mrow + v;         // m = b*T + t
        int t = m & (SEQT - 1);
        int b = m >> TLOG;
        size_t row = (size_t)t * BATCH + b;      // [T,B,*] layout
        Out[row * GATES + n0] = acc0[v] + bb0;
        Out[row * GATES + n1] = acc1[v] + bb1;
    }
}

// ---------------------------------------------------------------------------
// GEMM (f16 inputs, layer 1): staging via GLOBAL_LOAD_ASYNC_TO_LDS_B128.
// Each lane's fragment slot-run is 8 contiguous f16 = 16 bytes, so one async
// lane-transfer lands it directly in WMMA fragment order -- 2 async
// instructions per wave per K-panel, zero VGPR round-trip, ASYNCcnt-tracked.
// ---------------------------------------------------------------------------
__global__ __launch_bounds__(256) void wmma_gemm_bias_f16(
    const _Float16* __restrict__ A, const _Float16* __restrict__ W,
    const float* __restrict__ bias0, const float* __restrict__ bias1,
    float* __restrict__ Out, int K)
{
    __shared__ _Float16 sA[2][4 * 512];
    __shared__ _Float16 sB[2][4 * 512];

    const int tid  = threadIdx.x;
    const int wave = tid >> 5;
    const int lane = tid & 31;
    const int bm   = blockIdx.x * 64;
    const int bn   = blockIdx.y * 64;
    const int wm   = wave & 3;
    const int wn   = (wave >> 2) * 2;

    const int r    = tid >> 2;
    const int q    = tid & 3;
    const int hi   = (q & 1) ? 16 : 0;
    const int sb   = (q >> 1) * 8;
    const int ldso = (r >> 4) * 512 + ((r & 15) + hi) * 16 + sb;
    const _Float16* Ap = A + (size_t)(bm + r) * K + q * 8;
    const _Float16* Wp = W + (size_t)(bn + r) * K + q * 8;

    const uint32_t lA0 = lds_addr32(&sA[0][ldso]), lA1 = lds_addr32(&sA[1][ldso]);
    const uint32_t lB0 = lds_addr32(&sB[0][ldso]), lB1 = lds_addr32(&sB[1][ldso]);

    v8f acc0 = {}, acc1 = {};
    const int nIter = K >> 5;

    {   // prologue: async-stage panel 0
        uint64_t ga = (uint64_t)(uintptr_t)Ap;
        uint64_t gw = (uint64_t)(uintptr_t)Wp;
        asm volatile("global_load_async_to_lds_b128 %0, %1, off"
                     :: "v"(lA0), "v"(ga) : "memory");
        asm volatile("global_load_async_to_lds_b128 %0, %1, off"
                     :: "v"(lB0), "v"(gw) : "memory");
    }

    for (int i = 0; i < nIter; ++i) {
        asm volatile("s_wait_asynccnt 0x0" ::: "memory");  // my panel landed
        __syncthreads();                                   // everyone's landed
        const int cur = i & 1;
        if (i + 1 < nIter) {           // async-stage next panel into other buf
            uint64_t ga = (uint64_t)(uintptr_t)(Ap + (size_t)(i + 1) * 32);
            uint64_t gw = (uint64_t)(uintptr_t)(Wp + (size_t)(i + 1) * 32);
            asm volatile("global_load_async_to_lds_b128 %0, %1, off"
                         :: "v"(cur ? lA0 : lA1), "v"(ga) : "memory");
            asm volatile("global_load_async_to_lds_b128 %0, %1, off"
                         :: "v"(cur ? lB0 : lB1), "v"(gw) : "memory");
        }
        v16h af  = *(const v16h*)&sA[cur][wm * 512 + lane * 16];
        v16h bf0 = *(const v16h*)&sB[cur][(wn + 0) * 512 + lane * 16];
        v16h bf1 = *(const v16h*)&sB[cur][(wn + 1) * 512 + lane * 16];
        acc0 = __builtin_amdgcn_wmma_f32_16x16x32_f16(false, af, false, bf0,
                                                      (short)0, acc0, false, false);
        acc1 = __builtin_amdgcn_wmma_f32_16x16x32_f16(false, af, false, bf1,
                                                      (short)0, acc1, false, false);
    }

    const int mrow = (lane >= 16) ? 8 : 0;
    const int ncol = lane & 15;
    const int n0   = bn + wn * 16 + ncol;
    const int n1   = n0 + 16;
    const float bb0 = bias0[n0] + bias1[n0];
    const float bb1 = bias0[n1] + bias1[n1];
    #pragma unroll
    for (int v = 0; v < 8; ++v) {
        int m = bm + wm * 16 + mrow + v;
        int t = m & (SEQT - 1);
        int b = m >> TLOG;
        size_t row = (size_t)t * BATCH + b;
        Out[row * GATES + n0] = acc0[v] + bb0;
        Out[row * GATES + n1] = acc1[v] + bb1;
    }
}

// ---------------------------------------------------------------------------
// LSTM recurrence. Grid = B/16 (independent batch tiles), 256 thr / 8 waves.
// Wave w owns h columns [16w,16w+16) -> gate tiles {w,w+8,w+16,w+24}.
// w_hh in 128 VGPRs/wave for the whole T loop; h exchanged via 4 KB LDS
// fragment buffer; next step's xg loaded into a second register set during
// the current step's WMMAs. seq mode writes f16 h-sequence (feeds the async
// f16 GEMM of the next layer); last mode writes f32 final h.
// ---------------------------------------------------------------------------
__global__ __launch_bounds__(256) void lstm_recur(
    const float* __restrict__ xg,      // [T, B, 512] (bias folded in)
    const float* __restrict__ whh,     // [512, 128] fp32
    _Float16* __restrict__ hseq,       // [B*T, H] f16 (seq mode)
    float* __restrict__ hlast,         // [B, H] f32 (last mode)
    int T, int seq)
{
    __shared__ _Float16 sH[4 * 512];   // 4 K-fragments of A (16x32)

    const int wave = threadIdx.x >> 5;
    const int lane = threadIdx.x & 31;
    const int b0   = blockIdx.x * 16;
    const int mrow = (lane >= 16) ? 8 : 0;
    const int ncol = lane & 15;

    // ---- preload weight fragments (B layout: n on lanes, k in slots) ----
    v16h wf[4][4];
    #pragma unroll
    for (int g = 0; g < 4; ++g) {
        const int ng = g * HID + (wave << 4) + ncol;
        const float* wrow = whh + (size_t)ng * HID;
        #pragma unroll
        for (int kk = 0; kk < 4; ++kk) {
            const int kb = kk * 32 + ((lane >= 16) ? 8 : 0);
            f32x4 x0 = *(const f32x4*)(wrow + kb);
            f32x4 x1 = *(const f32x4*)(wrow + kb + 4);
            f32x4 y0 = *(const f32x4*)(wrow + kb + 16);
            f32x4 y1 = *(const f32x4*)(wrow + kb + 20);
            v8h lo = pack8(x0, x1), hiv = pack8(y0, y1);
            v16h f;
            #pragma unroll
            for (int s = 0; s < 8; ++s) { f[s] = lo[s]; f[8 + s] = hiv[s]; }
            wf[g][kk] = f;
        }
    }

    // h column this thread feeds back, and its A-fragment (kk, lane', slot):
    const int kcol = (wave << 4) + ncol;
    const int kk_w = kcol >> 5;
    const int kr   = kcol & 31;
    const int hi16 = ((kr & 15) >= 8) ? 16 : 0;
    const int ss   = (kr & 7) + ((kr >= 16) ? 8 : 0);
    _Float16* hdst = &sH[kk_w * 512 + ss];

    for (int i = threadIdx.x; i < 4 * 512; i += 256) sH[i] = (_Float16)0.f;
    v8f c = {};

    // ---- prime xg pipeline for t = 0 ----
    v8f nxt[4];
    #pragma unroll
    for (int g = 0; g < 4; ++g) {
        const int nc = g * HID + kcol;
        #pragma unroll
        for (int v = 0; v < 8; ++v)
            nxt[g][v] = xg[(size_t)(b0 + mrow + v) * GATES + nc];
    }
    __syncthreads();

    for (int t = 0; t < T; ++t) {
        v16h a[4];
        #pragma unroll
        for (int kk = 0; kk < 4; ++kk)
            a[kk] = *(const v16h*)&sH[kk * 512 + lane * 16];

        v8f acc[4];
        #pragma unroll
        for (int g = 0; g < 4; ++g) acc[g] = nxt[g];

        if (t + 1 < T) {               // next step's xg, hidden behind WMMAs
            #pragma unroll
            for (int g = 0; g < 4; ++g) {
                const int nc = g * HID + kcol;
                #pragma unroll
                for (int v = 0; v < 8; ++v)
                    nxt[g][v] = xg[((size_t)(t + 1) * BATCH + b0 + mrow + v) * GATES + nc];
            }
        }

        #pragma unroll
        for (int kk = 0; kk < 4; ++kk) {
            #pragma unroll
            for (int g = 0; g < 4; ++g)
                acc[g] = __builtin_amdgcn_wmma_f32_16x16x32_f16(
                    false, a[kk], false, wf[g][kk], (short)0, acc[g], false, false);
        }

        __syncthreads();   // all waves done reading sH for step t

        #pragma unroll
        for (int v = 0; v < 8; ++v) {
            float iv = fast_sigmoid(acc[0][v]);
            float fv = fast_sigmoid(acc[1][v]);
            float gv = fast_tanh(acc[2][v]);
            float ov = fast_sigmoid(acc[3][v]);
            float cv = fv * c[v] + iv * gv;
            c[v] = cv;
            float hv = ov * fast_tanh(cv);
            if (seq)
                hseq[((size_t)(b0 + mrow + v) * T + t) * HID + kcol] = (_Float16)hv;
            else
                hlast[(size_t)(b0 + mrow + v) * HID + kcol] = hv;
            hdst[(size_t)(mrow + v + hi16) * 16] = (_Float16)hv;
        }
        __syncthreads();   // new h visible before next step's reads
    }
}

// ---------------------------------------------------------------------------
__global__ __launch_bounds__(256) void cvt_f32_to_f16(
    const float* __restrict__ in, _Float16* __restrict__ outp, int n)
{
    int i = blockIdx.x * blockDim.x + threadIdx.x;
    if (i < n) outp[i] = (_Float16)in[i];
}

// Final FC: logits[128,8] = h2_last[128,128] * fc_w[8,128]^T + fc_b
__global__ __launch_bounds__(256) void fc_kernel(
    const float* __restrict__ h, const float* __restrict__ w,
    const float* __restrict__ b, float* __restrict__ out)
{
    int idx = blockIdx.x * blockDim.x + threadIdx.x;
    if (idx >= BATCH * 8) return;
    int bi = idx >> 3, j = idx & 7;
    float s = b[j];
    #pragma unroll 4
    for (int n = 0; n < HID; ++n) s += h[bi * HID + n] * w[j * HID + n];
    out[idx] = s;
}

extern "C" void kernel_launch(void* const* d_in, const int* in_sizes, int n_in,
                              void* d_out, int out_size, void* d_ws, size_t ws_size,
                              hipStream_t stream) {
    const float* x     = (const float*)d_in[0];
    const float* w_ih0 = (const float*)d_in[1];
    const float* w_hh0 = (const float*)d_in[2];
    const float* b_ih0 = (const float*)d_in[3];
    const float* b_hh0 = (const float*)d_in[4];
    const float* w_ih1 = (const float*)d_in[5];
    const float* w_hh1 = (const float*)d_in[6];
    const float* b_ih1 = (const float*)d_in[7];
    const float* b_hh1 = (const float*)d_in[8];
    const float* fc_w  = (const float*)d_in[9];
    const float* fc_b  = (const float*)d_in[10];
    float* out = (float*)d_out;

    char* ws = (char*)d_ws;
    const size_t XG_B = (size_t)MROWS * GATES * 4;   // 128 MB (reused)
    const size_t H1_B = (size_t)MROWS * HID * 2;     // 16 MB f16
    const size_t H2_B = (size_t)BATCH * HID * 4;     // 64 KB
    float*    xg  = (float*)ws;
    _Float16* h1h = (_Float16*)(ws + XG_B);
    float*    h2  = (float*)(ws + XG_B + H1_B);
    _Float16* w1h = (_Float16*)(ws + XG_B + H1_B + H2_B);  // 128 KB f16

    dim3 ggrid(MROWS / 64, GATES / 64);

    // w_ih1 -> f16 (independent; overlaps with layer-0 work)
    cvt_f32_to_f16<<<(GATES * HID + 255) / 256, 256, 0, stream>>>(w_ih1, w1h, GATES * HID);

    // Layer 0: fp32 GEMM (K=1024), then recurrence (writes f16 h1 sequence)
    wmma_gemm_bias_f32<<<ggrid, 256, 0, stream>>>(x, w_ih0, b_ih0, b_hh0, xg, 1024);
    lstm_recur<<<BATCH / 16, 256, 0, stream>>>(xg, w_hh0, h1h, nullptr, SEQT, 1);

    // Layer 1: f16 GEMM with async-to-LDS staging (K=128), recurrence keeps last h
    wmma_gemm_bias_f16<<<ggrid, 256, 0, stream>>>(h1h, w1h, b_ih1, b_hh1, xg, HID);
    lstm_recur<<<BATCH / 16, 256, 0, stream>>>(xg, w_hh1, nullptr, h2, SEQT, 0);

    // Classifier head
    fc_kernel<<<4, 256, 0, stream>>>(h2, fc_w, fc_b, out);
}
